// NSFormer_49615462203755
// MI455X (gfx1250) — matrix operand order repaired and checked by reading, not compiled, not run on hardware
//
#include <hip/hip_runtime.h>
#include <cstdint>

// ---------------------------------------------------------------------------
// CDNA5 / gfx1250 NSFormer forward.  All matmuls go through
// v_wmma_f32_16x16x32_f16 (f16 in, f32 accumulate).  wave32, 256-thr blocks.
// Global->LDS staging in the fast GEMM uses gfx1250 async-to-LDS DMA.
// ---------------------------------------------------------------------------

typedef __attribute__((ext_vector_type(16))) _Float16 v16h;
typedef __attribute__((ext_vector_type(8)))  float    v8f;
typedef __attribute__((ext_vector_type(8)))  _Float16 v8h;
typedef __attribute__((ext_vector_type(4)))  _Float16 v4h;
typedef __attribute__((ext_vector_type(2)))  _Float16 v2h;
typedef _Float16 half_t;

// Model constants
static constexpr int kB  = 8;     // batch
static constexpr int kL  = 1024;  // seq len (enc and dec)
static constexpr int kD  = 512;   // d_model
static constexpr int kH  = 8;     // heads
static constexpr int kE  = 64;    // head dim
static constexpr int kF  = 2048;  // d_ff
static constexpr int kC  = 32;    // enc_in / c_out
static constexpr int kMk = 4;     // time marks
static constexpr long kBL  = (long)kB * kL;         // 8192
static constexpr long kBLD = kBL * kD;              // 4194304

// gfx1250 async global->LDS DMA (ASYNCcnt-tracked).  Low 32 bits of a generic
// LDS pointer are the LDS byte offset (ISA 10.2 aperture mapping).
__device__ __forceinline__ void gl_async_b128(void* lds, const void* gptr) {
  uint32_t l = (uint32_t)(uintptr_t)lds;
  asm volatile("global_load_async_to_lds_b128 %0, %1, off"
               :: "v"(l), "v"(gptr) : "memory");
}
__device__ __forceinline__ void wait_async0() {
  asm volatile("s_wait_asynccnt 0x0" ::: "memory");
}

// ===========================================================================
// WMMA GEMM:  C[M,N] = act(A[M,K] * B[K,N] + bias)
//   A f16, row stride lda.  B f16, element (k,n) at k*ldb_k + n*ldb_n.
//   Optional f32 out Cf and/or f16 out Ch (row stride ldc).
//   Batched over blockIdx.z with per-z element offsets offA/offB/offC.
// ===========================================================================
struct GemmArgs {
  const half_t* A; const half_t* B; const float* bias;
  float* Cf; half_t* Ch;
  int M, N, K, lda, ldb_k, ldb_n, ldc, relu;
  long offA, offB, offC;
};

// ---------------------------------------------------------------------------
// Fast path: M%256==0, N%64==0, K%32==0.  Block tile 256x64, 8 waves, each
// wave owns a 32x64 tile (2x4 WMMA frags -> 8 v_wmma per K-chunk per wave).
// Double-buffered LDS; tiles prefetched with global_load_async_to_lds_b128.
// BMODE 0: ldb_n==1 (contiguous along n; B register-transposed into Bs[n][k])
// BMODE 1: ldb_k==1 (contiguous along k; B async-DMA direct into Bs[n][k])
// ---------------------------------------------------------------------------
template<int BMODE>
__global__ __launch_bounds__(256)
void k_wmma_gemm_fast(GemmArgs g) {
  __shared__ half_t As[2][256][40];
  __shared__ half_t Bs[2][64][40];
  const int tid = threadIdx.x, lane = tid & 31, wave = tid >> 5;
  const int z = blockIdx.z;
  const half_t* Ag = g.A + (long)z * g.offA + (long)(blockIdx.y * 256) * g.lda;
  const half_t* Bg = g.B + (long)z * g.offB;
  const int n0 = blockIdx.x * 64;
  const long coff = (long)z * g.offC;

  // staging coords
  const int ai = tid >> 2;              // A rows (step 64 per it)
  const int aj = (tid & 3) << 3;        // A col within 32-chunk
  const int bk = (BMODE == 0) ? (tid >> 3) : ((tid & 3) << 3);
  const int bn = (BMODE == 0) ? ((tid & 7) << 3) : (tid >> 2);

  v8h breg;                             // BMODE 0 B staging only
  auto prefetch = [&](int k0, int buf) {
#pragma unroll
    for (int it = 0; it < 4; ++it)
      gl_async_b128(&As[buf][ai + it * 64][aj],
                    Ag + (long)(ai + it * 64) * g.lda + k0 + aj);
    if (BMODE == 1)
      gl_async_b128(&Bs[buf][bn][bk],
                    Bg + (long)(n0 + bn) * g.ldb_n + (k0 + bk));
    else
      breg = *(const v8h*)(Bg + (long)(k0 + bk) * g.ldb_k + (n0 + bn));
  };
  auto commitB = [&](int buf) {
    if (BMODE == 0) {
#pragma unroll
      for (int e = 0; e < 8; ++e) Bs[buf][bn + e][bk] = breg[e];
    }
  };

  v8f acc[2][4] = {};
  const int ml = lane & 15;
  const int ahalf = (lane >> 4) << 3;   // A: lanes 16-31 hold K+8 / K+24
  const int bhalf = (lane >> 4) << 4;   // B: lanes 16-31 hold K+8 / K+24 (bhalf>>1)
  const int nk = g.K >> 5;

  prefetch(0, 0);
  commitB(0);
  wait_async0();
  __syncthreads();
  for (int t = 0; t < nk; ++t) {
    const int cur = t & 1;
    if (t + 1 < nk) prefetch((t + 1) << 5, cur ^ 1);  // DMA next tile into LDS
    v16h af[2], bf[4];
#pragma unroll
    for (int i = 0; i < 2; ++i) {
      const half_t* ar = &As[cur][wave * 32 + i * 16 + ml][0];
#pragma unroll
      for (int v = 0; v < 8; ++v) {
        int kk = ((v >> 2) << 4) + ahalf + ((v & 3) << 1);
        v2h p = *(const v2h*)&ar[kk];
        af[i][2 * v] = p.x; af[i][2 * v + 1] = p.y;
      }
    }
#pragma unroll
    for (int j = 0; j < 4; ++j) {
      const half_t* br = &Bs[cur][j * 16 + ml][0];
#pragma unroll
      for (int v = 0; v < 8; ++v) {
        int kk = ((v >> 2) << 4) + (bhalf >> 1) + ((v & 3) << 1);
        v2h p = *(const v2h*)&br[kk];
        bf[j][2 * v] = p.x; bf[j][2 * v + 1] = p.y;
      }
    }
#pragma unroll
    for (int i = 0; i < 2; ++i)
#pragma unroll
      for (int j = 0; j < 4; ++j)
        acc[i][j] = __builtin_amdgcn_wmma_f32_16x16x32_f16(
            false, af[i], false, bf[j], (short)0, acc[i][j], false, false);
    if (t + 1 < nk) { commitB(cur ^ 1); wait_async0(); }
    __syncthreads();
  }

  // epilogue: VGPR r holds row r (lanes 0-15) / r+8 (lanes 16-31), col lane%16
  const int m0 = blockIdx.y * 256 + wave * 32;
  const int rsel = (lane >> 4) << 3;
#pragma unroll
  for (int i = 0; i < 2; ++i)
#pragma unroll
    for (int j = 0; j < 4; ++j) {
      int nc = n0 + j * 16 + ml;
      float bv = g.bias ? g.bias[nc] : 0.f;
#pragma unroll
      for (int r = 0; r < 8; ++r) {
        int mr = m0 + i * 16 + rsel + r;
        float x = acc[i][j][r] + bv;
        if (g.relu) x = fmaxf(x, 0.f);
        long idx = coff + (long)mr * g.ldc + nc;
        if (g.Cf) g.Cf[idx] = x;
        if (g.Ch) g.Ch[idx] = (half_t)x;
      }
    }
}

// ---------------------------------------------------------------------------
// Generic guarded path (any M/N; K%32==0).  Block tile 128x64, 8 waves.
// Used only for odd shapes (final 512->32 projection).
// ---------------------------------------------------------------------------
__global__ __launch_bounds__(256)
void k_wmma_gemm_gen(GemmArgs g) {
  __shared__ half_t As[128][40];
  __shared__ half_t Bs[64][40];
  const int tid  = threadIdx.x;
  const int lane = tid & 31;
  const int wave = tid >> 5;
  const int wm = wave >> 1;
  const int wn = wave & 1;
  const int z = blockIdx.z;
  const half_t* A = g.A + (long)z * g.offA;
  const half_t* B = g.B + (long)z * g.offB;
  const long coff = (long)z * g.offC;
  const int m0 = blockIdx.y * 128;
  const int n0 = blockIdx.x * 64;

  v8f acc[2][2] = {};

  for (int k0 = 0; k0 < g.K; k0 += 32) {
#pragma unroll
    for (int it = 0; it < 4; ++it) {
      int c = tid + (it << 8);
      int i = c >> 3;
      int j = (c & 7) << 2;
      v4h val = {};
      int m = m0 + i;
      if (m < g.M) val = *(const v4h*)(A + (long)m * g.lda + (k0 + j));
      *(v4h*)&As[i][j] = val;
    }
#pragma unroll
    for (int it = 0; it < 8; ++it) {
      int c = tid + (it << 8);
      int n = c >> 5;
      int k = c & 31;
      half_t val = (half_t)0.f;
      if (n0 + n < g.N)
        val = B[(long)(k0 + k) * g.ldb_k + (long)(n0 + n) * g.ldb_n];
      Bs[n][k] = val;
    }
    __syncthreads();

    const int ml = lane & 15;
    const int ahalf = (lane >> 4) << 3;
    const int bhalf = (lane >> 4) << 4;
    v16h af[2], bf[2];
#pragma unroll
    for (int i = 0; i < 2; ++i) {
      const half_t* ar = &As[wm * 32 + i * 16 + ml][0];
#pragma unroll
      for (int v = 0; v < 8; ++v) {
        int kk = ((v >> 2) << 4) + ahalf + ((v & 3) << 1);
        v2h p = *(const v2h*)&ar[kk];
        af[i][2 * v] = p.x; af[i][2 * v + 1] = p.y;
      }
    }
#pragma unroll
    for (int j = 0; j < 2; ++j) {
      const half_t* br = &Bs[wn * 32 + j * 16 + ml][0];
#pragma unroll
      for (int v = 0; v < 8; ++v) {
        int kk = ((v >> 2) << 4) + (bhalf >> 1) + ((v & 3) << 1);
        v2h p = *(const v2h*)&br[kk];
        bf[j][2 * v] = p.x; bf[j][2 * v + 1] = p.y;
      }
    }
#pragma unroll
    for (int i = 0; i < 2; ++i)
#pragma unroll
      for (int j = 0; j < 2; ++j)
        acc[i][j] = __builtin_amdgcn_wmma_f32_16x16x32_f16(
            false, af[i], false, bf[j], (short)0, acc[i][j], false, false);
    __syncthreads();
  }

  const int nl = lane & 15;
  const int rsel = (lane >> 4) << 3;
#pragma unroll
  for (int i = 0; i < 2; ++i)
#pragma unroll
    for (int j = 0; j < 2; ++j) {
      int nc = n0 + wn * 32 + j * 16 + nl;
      if (nc >= g.N) continue;
      float bv = g.bias ? g.bias[nc] : 0.f;
#pragma unroll
      for (int r = 0; r < 8; ++r) {
        int mr = m0 + wm * 32 + i * 16 + rsel + r;
        if (mr >= g.M) continue;
        float x = acc[i][j][r] + bv;
        if (g.relu) x = fmaxf(x, 0.f);
        long idx = coff + (long)mr * g.ldc + nc;
        if (g.Cf) g.Cf[idx] = x;
        if (g.Ch) g.Ch[idx] = (half_t)x;
      }
    }
}

// ===========================================================================
// Elementwise / reduction kernels (f32 path)
// ===========================================================================
__global__ void k_cvt(const float* s, half_t* d, long n) {
  long i = (long)blockIdx.x * 256 + threadIdx.x;
  if (i < n) d[i] = (half_t)s[i];
}

__global__ void k_copy(const float* s, float* d, long n) {
  long i = (long)blockIdx.x * 256 + threadIdx.x;
  if (i < n) d[i] = s[i];
}

// per-(b,c) mean / std over seq axis
__global__ __launch_bounds__(256)
void k_stats(const float* x, float* mean, float* stdv) {
  __shared__ float r1[256], r2[256];
  int bc = blockIdx.x;                  // b*32 + c
  int b = bc >> 5, c = bc & 31;
  float s = 0.f, s2 = 0.f;
  for (int i = 0; i < 4; ++i) {
    int l = threadIdx.x + (i << 8);
    float v = x[((long)b * kL + l) * kC + c];
    s += v; s2 += v * v;
  }
  r1[threadIdx.x] = s; r2[threadIdx.x] = s2; __syncthreads();
  for (int o = 128; o; o >>= 1) {
    if (threadIdx.x < o) { r1[threadIdx.x] += r1[threadIdx.x + o];
                           r2[threadIdx.x] += r2[threadIdx.x + o]; }
    __syncthreads();
  }
  if (threadIdx.x == 0) {
    float m = r1[0] * (1.f / kL);
    float var = r2[0] * (1.f / kL) - m * m;
    mean[bc] = m;
    stdv[bc] = sqrtf(var + 1e-5f);
  }
}

__global__ void k_norm(const float* x, const float* mean, const float* stdv, float* xn) {
  long i = (long)blockIdx.x * 256 + threadIdx.x;
  if (i >= (long)kB * kL * kC) return;
  int c = i & 31;
  long bl = i >> 5;
  int b = (int)(bl / kL);
  xn[i] = (x[i] - mean[b * kC + c]) / stdv[b * kC + c];
}

// x_dec_new = concat(xn[:, -512:, :], zeros)
__global__ void k_build_xdec(const float* xn, float* xd) {
  long i = (long)blockIdx.x * 256 + threadIdx.x;
  if (i >= (long)kB * kL * kC) return;
  int c = i & 31;
  long bl = i >> 5;
  int l = (int)(bl % kL);
  int b = (int)(bl / kL);
  xd[i] = (l < 512) ? xn[((long)b * kL + 512 + l) * kC + c] : 0.f;
}

// token circular conv(k=3) + temporal linear + sinusoidal PE
__global__ void k_embed(const float* x, const float* mark, const float* tokw,
                        const float* tempw, float* Of, half_t* Oh) {
  long idx = (long)blockIdx.x * 256 + threadIdx.x;
  if (idx >= kBLD) return;
  int d = (int)(idx & (kD - 1));
  long bl = idx >> 9;
  int l = (int)(bl % kL);
  int b = (int)(bl / kL);
  float acc = 0.f;
  for (int kk = 0; kk < 3; ++kk) {
    int ls = (l + kk - 1 + kL) % kL;
    const float* xr = x + ((long)b * kL + ls) * kC;
    const float* wr = tokw + (long)d * kC * 3 + kk;   // tokw[d][c][kk]
    for (int c = 0; c < kC; ++c) acc += wr[c * 3] * xr[c];
  }
  const float* mr = mark + ((long)b * kL + l) * kMk;
  for (int mk = 0; mk < kMk; ++mk) acc += mr[mk] * tempw[mk * kD + d];
  int i2 = (d >> 1) << 1;
  float div = expf((float)i2 * (-9.210340371976184f / (float)kD));
  float ang = (float)l * div;
  acc += (d & 1) ? cosf(ang) : sinf(ang);
  Of[idx] = acc;
  Oh[idx] = (half_t)acc;
}

// tau/delta projector: series conv (1024ch -> 1 over width 32) + 2-hidden MLP
__global__ __launch_bounds__(128)
void k_projector(const float* xraw, const float* stats, const float* convw,
                 const float* w0, const float* b0, const float* w1, const float* b1,
                 const float* w2, float* out, int outdim, int do_exp) {
  __shared__ float h0[64], h1[128], h2[128];
  int b = blockIdx.x, t = threadIdx.x;
  if (t < kC) {
    float acc = 0.f;
    int jm1 = (t + kC - 1) % kC, jp1 = (t + 1) % kC;
    for (int c = 0; c < kL; ++c) {
      const float* xr = xraw + ((long)b * kL + c) * kC;
      const float* wr = convw + c * 3;
      acc += wr[0] * xr[jm1] + wr[1] * xr[t] + wr[2] * xr[jp1];
    }
    h0[t] = acc;
  } else if (t < 2 * kC) {
    h0[t] = stats[b * kC + (t - kC)];
  }
  __syncthreads();
  { float acc = b0[t];
    for (int i = 0; i < 64; ++i) acc += h0[i] * w0[i * 128 + t];
    h1[t] = fmaxf(acc, 0.f); }
  __syncthreads();
  { float acc = b1[t];
    for (int i = 0; i < 128; ++i) acc += h1[i] * w1[i * 128 + t];
    h2[t] = fmaxf(acc, 0.f); }
  __syncthreads();
  for (int o = t; o < outdim; o += 128) {
    float acc = 0.f;
    for (int i = 0; i < 128; ++i) acc += h2[i] * w2[i * outdim + o];
    out[(long)b * outdim + o] = do_exp ? expf(acc) : acc;
  }
}

// softmax((tau*s + delta) * 1/sqrt(E)) with optional causal mask, f16 probs out
__global__ __launch_bounds__(256)
void k_softmax(const float* S, half_t* P, const float* tau, const float* delta,
               int L, int Slen, int causal) {
  __shared__ float red[256];
  int row = blockIdx.x;                 // h*L + l
  int l = row % L;
  const float* sr = S + (long)row * Slen;
  half_t* pr = P + (long)row * Slen;
  float t = tau[0];
  float v[8];
  int per = (Slen + 255) >> 8;
  float mx = -3.0e38f;
  for (int i = 0; i < per; ++i) {
    int s = threadIdx.x + (i << 8);
    float x = -3.0e38f;
    if (s < Slen && (!causal || s <= l)) {
      x = t * sr[s];
      if (delta) x += delta[s];
      x *= 0.125f;                      // 1/sqrt(64)
    }
    v[i] = x;
    mx = fmaxf(mx, x);
  }
  red[threadIdx.x] = mx; __syncthreads();
  for (int o = 128; o; o >>= 1) {
    if (threadIdx.x < o) red[threadIdx.x] = fmaxf(red[threadIdx.x], red[threadIdx.x + o]);
    __syncthreads();
  }
  mx = red[0]; __syncthreads();
  float sum = 0.f;
  for (int i = 0; i < per; ++i) {
    float e = (v[i] > -2.9e38f) ? expf(v[i] - mx) : 0.f;
    v[i] = e; sum += e;
  }
  red[threadIdx.x] = sum; __syncthreads();
  for (int o = 128; o; o >>= 1) {
    if (threadIdx.x < o) red[threadIdx.x] += red[threadIdx.x + o];
    __syncthreads();
  }
  float inv = 1.f / red[0];
  for (int i = 0; i < per; ++i) {
    int s = threadIdx.x + (i << 8);
    if (s < Slen) pr[s] = (half_t)(v[i] * inv);
  }
}

// out = LN(x (+ y)) * gamma + beta  (D=512, one row per block), f32 + f16 out
__global__ __launch_bounds__(256)
void k_add_ln(const float* X, const float* Y, const float* gamma, const float* beta,
              float* Of, half_t* Oh) {
  __shared__ float red[256];
  long row = blockIdx.x;
  const float* xr = X + row * kD;
  const float* yr = Y ? Y + row * kD : nullptr;
  int i0 = threadIdx.x, i1 = threadIdx.x + 256;
  float v0 = xr[i0] + (yr ? yr[i0] : 0.f);
  float v1 = xr[i1] + (yr ? yr[i1] : 0.f);
  red[threadIdx.x] = v0 + v1; __syncthreads();
  for (int o = 128; o; o >>= 1) {
    if (threadIdx.x < o) red[threadIdx.x] += red[threadIdx.x + o];
    __syncthreads();
  }
  float mean = red[0] * (1.f / kD); __syncthreads();
  float d0 = v0 - mean, d1 = v1 - mean;
  red[threadIdx.x] = d0 * d0 + d1 * d1; __syncthreads();
  for (int o = 128; o; o >>= 1) {
    if (threadIdx.x < o) red[threadIdx.x] += red[threadIdx.x + o];
    __syncthreads();
  }
  float rs = rsqrtf(red[0] * (1.f / kD) + 1e-5f);
  float o0 = d0 * rs * gamma[i0] + beta[i0];
  float o1 = d1 * rs * gamma[i1] + beta[i1];
  Of[row * kD + i0] = o0; Of[row * kD + i1] = o1;
  if (Oh) { Oh[row * kD + i0] = (half_t)o0; Oh[row * kD + i1] = (half_t)o1; }
}

__global__ __launch_bounds__(256)
void k_kl_partial(const float* m, const float* lv, float* part, long n) {
  __shared__ float red[256];
  float s = 0.f;
  for (long i = (long)blockIdx.x * 256 + threadIdx.x; i < n; i += (long)gridDim.x * 256) {
    float mm = m[i], l = lv[i];
    s += 1.f - mm * mm + l - expf(l);
  }
  red[threadIdx.x] = s; __syncthreads();
  for (int o = 128; o; o >>= 1) {
    if (threadIdx.x < o) red[threadIdx.x] += red[threadIdx.x + o];
    __syncthreads();
  }
  if (threadIdx.x == 0) part[blockIdx.x] = red[0];
}

__global__ __launch_bounds__(256)
void k_kl_final(const float* part, float* out, int nb, float scale) {
  __shared__ float red[256];
  float s = 0.f;
  for (int i = threadIdx.x; i < nb; i += 256) s += part[i];
  red[threadIdx.x] = s; __syncthreads();
  for (int o = 128; o; o >>= 1) {
    if (threadIdx.x < o) red[threadIdx.x] += red[threadIdx.x + o];
    __syncthreads();
  }
  if (threadIdx.x == 0) out[0] = -0.5f * red[0] * scale;
}

// dec_out = proj * std + mean; write full + pred slice into d_out
__global__ void k_finalize(const float* dp, const float* mean, const float* stdv,
                           float* out_pred, float* out_full) {
  long i = (long)blockIdx.x * 256 + threadIdx.x;
  if (i >= (long)kB * kL * kC) return;
  int c = i & 31;
  long bl = i >> 5;
  int l = (int)(bl % kL);
  int b = (int)(bl / kL);
  float v = dp[i] * stdv[b * kC + c] + mean[b * kC + c];
  out_full[i] = v;
  if (l >= 512)
    out_pred[((long)b * 512 + (l - 512)) * kC + c] = v;
}

// ===========================================================================
// Host orchestration
// ===========================================================================
static inline void gemm(hipStream_t st, const half_t* A, const half_t* B,
                        const float* bias, float* Cf, half_t* Ch,
                        int M, int N, int K, int lda, int ldb_k, int ldb_n, int ldc,
                        int Z, long offA, long offB, long offC, int relu) {
  GemmArgs g{A, B, bias, Cf, Ch, M, N, K, lda, ldb_k, ldb_n, ldc, relu, offA, offB, offC};
  bool full = (M % 256 == 0) && (N % 64 == 0) && (K % 32 == 0);
  if (full) {
    dim3 gr(N / 64, M / 256, Z);
    if (ldb_n == 1) k_wmma_gemm_fast<0><<<gr, dim3(256), 0, st>>>(g);
    else            k_wmma_gemm_fast<1><<<gr, dim3(256), 0, st>>>(g);
  } else {
    dim3 gr((N + 63) / 64, (M + 127) / 128, Z);
    k_wmma_gemm_gen<<<gr, dim3(256), 0, st>>>(g);
  }
}

// per-batch attention: scores = Q Kt, softmax, ctx = P V (heads batched in z)
static void attention(hipStream_t st, const half_t* q, const half_t* k,
                      const half_t* v, float* scores, half_t* probs, half_t* ctx,
                      const float* tau, const float* delta, int causal) {
  for (int b = 0; b < kB; ++b) {
    const half_t* qb = q + (long)b * kL * kD;
    const half_t* kb = k + (long)b * kL * kD;
    const half_t* vb = v + (long)b * kL * kD;
    // scores[h][l][s] = sum_e q[l, h*E+e] * k[s, h*E+e]
    gemm(st, qb, kb, nullptr, scores, nullptr,
         kL, kL, kE, kD, /*ldb_k=*/1, /*ldb_n=*/kD, /*ldc=*/kL,
         kH, /*offA=*/kE, /*offB=*/kE, /*offC=*/(long)kL * kL, 0);
    k_softmax<<<dim3(kH * kL), dim3(256), 0, st>>>(
        scores, probs, tau + b, delta ? delta + (long)b * kL : nullptr,
        kL, kL, causal);
    // ctx[l, h*E+e] = sum_s P[h][l][s] * v[s, h*E+e]
    gemm(st, probs, vb, nullptr, nullptr, ctx + (long)b * kL * kD,
         kL, kE, kL, /*lda=*/kL, /*ldb_k=*/kD, /*ldb_n=*/1, /*ldc=*/kD,
         kH, /*offA=*/(long)kL * kL, /*offB=*/kE, /*offC=*/kE, 0);
  }
}

extern "C" void kernel_launch(void* const* d_in, const int* in_sizes, int n_in,
                              void* d_out, int out_size, void* d_ws, size_t ws_size,
                              hipStream_t stream) {
  (void)in_sizes; (void)n_in; (void)out_size; (void)ws_size;
  auto F = [&](int i) { return (const float*)d_in[i]; };
  // --- inputs (setup_inputs insertion order) ---
  const float* x_enc      = F(0);
  const float* x_mark_enc = F(1);
  (void)F(2); // x_dec: only its zero-extension of xn is used
  const float* x_mark_dec = F(3);

  // --- params flattened in JAX pytree order (dict keys sorted recursively),
  // starting at d_in[4].  Top-level sorted: dec_b, dec_g, dec_layers, delta,
  // emb_dec, emb_enc, enc_b, enc_g, enc_layers, proj_b, proj_w, tau,
  // z_logvar, z_mean, z_out. ---
  const float* dec_b = F(4); const float* dec_g = F(5);
  const float *dl_b1n=F(6), *dl_b2n=F(7), *dl_b3n=F(8), *dl_bf1=F(9), *dl_bf2=F(10);
  const float *dc_bk=F(11), *dc_bo=F(12), *dc_bq=F(13), *dc_bv=F(14),
              *dc_wk=F(15), *dc_wo=F(16), *dc_wq=F(17), *dc_wv=F(18);
  const float *dl_g1=F(19), *dl_g2=F(20), *dl_g3=F(21);
  const float *ds_bk=F(22), *ds_bo=F(23), *ds_bq=F(24), *ds_bv=F(25),
              *ds_wk=F(26), *ds_wo=F(27), *ds_wq=F(28), *ds_wv=F(29);
  const float *dl_w1=F(30), *dl_w2=F(31);
  const float *pd_b0=F(32), *pd_b1=F(33), *pd_conv=F(34), *pd_w0=F(35),
              *pd_w1=F(36), *pd_w2=F(37);
  const float *embd_temp=F(38), *embd_tok=F(39);
  const float *embe_temp=F(40), *embe_tok=F(41);
  const float *enc_b=F(42), *enc_g=F(43);
  struct ELW { const float *bk,*bo,*bq,*bv,*wk,*wo,*wq,*wv,
                           *b1n,*b2n,*bf1,*bf2,*g1,*g2,*w1,*w2; } el[2];
  for (int l = 0; l < 2; ++l) {
    int o = 44 + l * 16;
    el[l] = { F(o+0),F(o+1),F(o+2),F(o+3),F(o+4),F(o+5),F(o+6),F(o+7),
              F(o+8),F(o+9),F(o+10),F(o+11),F(o+12),F(o+13),F(o+14),F(o+15) };
  }
  const float *proj_b=F(76), *proj_w=F(77);
  const float *pt_b0=F(78), *pt_b1=F(79), *pt_conv=F(80), *pt_w0=F(81),
              *pt_w1=F(82), *pt_w2=F(83);
  const float *zl_b1=F(84), *zl_b2=F(85), *zl_w1=F(86), *zl_w2=F(87);
  const float *zm_b1=F(88), *zm_b2=F(89), *zm_w1=F(90), *zm_w2=F(91);
  const float *zo_b1=F(92), *zo_b2=F(93), *zo_w1=F(94), *zo_w2=F(95);

  // --- workspace bump allocator ---
  char* wp = (char*)d_ws;
  auto alloc  = [&](size_t bytes) { char* r = wp; wp += (bytes + 255) & ~(size_t)255; return r; };
  auto allocH = [&](long n) { return (half_t*)alloc((size_t)n * sizeof(half_t)); };
  auto allocF = [&](long n) { return (float*)alloc((size_t)n * sizeof(float)); };

  // --- weight f16 conversion jobs ---
  struct CvtJob { const float* s; half_t* d; long n; } jobs[40]; int nj = 0;
  auto cvt = [&](const float* s, long n) {
    half_t* d = allocH(n); jobs[nj++] = {s, d, n}; return d;
  };
  const long WDD = (long)kD * kD, WDF = (long)kD * kF, WFD = (long)kF * kD;
  half_t *ewq[2], *ewk[2], *ewv[2], *ewo[2], *ew1[2], *ew2[2];
  for (int l = 0; l < 2; ++l) {
    ewq[l] = cvt(el[l].wq, WDD); ewk[l] = cvt(el[l].wk, WDD);
    ewv[l] = cvt(el[l].wv, WDD); ewo[l] = cvt(el[l].wo, WDD);
    ew1[l] = cvt(el[l].w1, WDF); ew2[l] = cvt(el[l].w2, WFD);
  }
  half_t *swq = cvt(ds_wq, WDD), *swk = cvt(ds_wk, WDD),
         *swv = cvt(ds_wv, WDD), *swo = cvt(ds_wo, WDD);
  half_t *cwq = cvt(dc_wq, WDD), *cwk = cvt(dc_wk, WDD),
         *cwv = cvt(dc_wv, WDD), *cwo = cvt(dc_wo, WDD);
  half_t *dw1 = cvt(dl_w1, WDF), *dw2 = cvt(dl_w2, WFD);
  half_t *zmw1 = cvt(zm_w1, WDD), *zmw2 = cvt(zm_w2, WDD);
  half_t *zlw1 = cvt(zl_w1, WDD), *zlw2 = cvt(zl_w2, WDD);
  half_t *zow1 = cvt(zo_w1, WDD), *zow2 = cvt(zo_w2, WDD);
  half_t *pw16 = cvt(proj_w, (long)kD * kC);

  // --- activations ---
  float*  h32   = allocF(kBLD);  half_t* h16  = allocH(kBLD);
  float*  dv32  = allocF(kBLD);  half_t* dv16 = allocH(kBLD);
  half_t* q16   = allocH(kBLD);
  half_t* kk16  = allocH(kBLD);
  half_t* vv16  = allocH(kBLD);
  half_t* ctx16 = allocH(kBLD);
  half_t* mid16 = allocH(kBL * (long)kF);
  float*  tmp32 = allocF(kBLD);
  float*  mean32 = allocF(kBLD); half_t* mean16 = allocH(kBLD);
  float*  logv32 = allocF(kBLD);
  half_t* t16   = allocH(kBLD);
  half_t* eo16  = allocH(kBLD);
  float*  scores = allocF((long)kH * kL * kL);
  half_t* probs  = allocH((long)kH * kL * kL);
  float*  xn32  = allocF((long)kB * kL * kC);
  float*  xd32  = allocF((long)kB * kL * kC);
  float*  dp32  = allocF(kBL * (long)kC);
  float*  tau32 = allocF(kB);
  float*  delta32 = allocF((long)kB * kL);
  float*  mean_s = allocF(kB * kC);
  float*  std_s  = allocF(kB * kC);
  float*  klpart = allocF(256);

  float* out = (float*)d_out;
  float* out_pred = out;                       // [8, 512, 32]
  float* out_full = out + 131072;              // [8, 1024, 32]
  float* out_kl   = out + 393216;              // scalar
  float* out_z    = out + 393217;              // [8, 1024, 512]

  // === convert weights ===
  for (int j = 0; j < nj; ++j)
    k_cvt<<<dim3((unsigned)((jobs[j].n + 255) / 256)), dim3(256), 0, stream>>>(
        jobs[j].s, jobs[j].d, jobs[j].n);

  // === stats / normalization / projectors ===
  k_stats<<<dim3(kB * kC), dim3(256), 0, stream>>>(x_enc, mean_s, std_s);
  long nBLC = (long)kB * kL * kC;
  k_norm<<<dim3((unsigned)((nBLC + 255) / 256)), dim3(256), 0, stream>>>(x_enc, mean_s, std_s, xn32);
  k_projector<<<dim3(kB), dim3(128), 0, stream>>>(
      x_enc, std_s, pt_conv, pt_w0, pt_b0, pt_w1, pt_b1, pt_w2, tau32, 1, 1);
  k_projector<<<dim3(kB), dim3(128), 0, stream>>>(
      x_enc, mean_s, pd_conv, pd_w0, pd_b0, pd_w1, pd_b1, pd_w2, delta32, kL, 0);

  unsigned gBLD = (unsigned)((kBLD + 255) / 256);

  // === encoder ===
  k_embed<<<dim3(gBLD), dim3(256), 0, stream>>>(xn32, x_mark_enc, embe_tok, embe_temp, h32, h16);
  for (int l = 0; l < 2; ++l) {
    gemm(stream, h16, ewq[l], el[l].bq, nullptr, q16,  (int)kBL, kD, kD, kD, kD, 1, kD, 1,0,0,0, 0);
    gemm(stream, h16, ewk[l], el[l].bk, nullptr, kk16, (int)kBL, kD, kD, kD, kD, 1, kD, 1,0,0,0, 0);
    gemm(stream, h16, ewv[l], el[l].bv, nullptr, vv16, (int)kBL, kD, kD, kD, kD, 1, kD, 1,0,0,0, 0);
    attention(stream, q16, kk16, vv16, scores, probs, ctx16, tau32, delta32, /*causal=*/0);
    gemm(stream, ctx16, ewo[l], el[l].bo, tmp32, nullptr, (int)kBL, kD, kD, kD, kD, 1, kD, 1,0,0,0, 0);
    k_add_ln<<<dim3((unsigned)kBL), dim3(256), 0, stream>>>(h32, tmp32, el[l].g1, el[l].b1n, h32, h16);
    gemm(stream, h16,  ew1[l], el[l].bf1, nullptr, mid16, (int)kBL, kF, kD, kD, kF, 1, kF, 1,0,0,0, 1);
    gemm(stream, mid16, ew2[l], el[l].bf2, tmp32, nullptr, (int)kBL, kD, kF, kF, kD, 1, kD, 1,0,0,0, 0);
    k_add_ln<<<dim3((unsigned)kBL), dim3(256), 0, stream>>>(h32, tmp32, el[l].g2, el[l].b2n, h32, h16);
  }
  k_add_ln<<<dim3((unsigned)kBL), dim3(256), 0, stream>>>(h32, nullptr, enc_g, enc_b, h32, h16);

  // === latent z path ===
  gemm(stream, h16, zmw1, zm_b1, nullptr, t16,    (int)kBL, kD, kD, kD, kD, 1, kD, 1,0,0,0, 1);
  gemm(stream, t16, zmw2, zm_b2, mean32, mean16,  (int)kBL, kD, kD, kD, kD, 1, kD, 1,0,0,0, 0);
  gemm(stream, h16, zlw1, zl_b1, nullptr, t16,    (int)kBL, kD, kD, kD, kD, 1, kD, 1,0,0,0, 1);
  gemm(stream, t16, zlw2, zl_b2, logv32, nullptr, (int)kBL, kD, kD, kD, kD, 1, kD, 1,0,0,0, 0);
  gemm(stream, mean16, zow1, zo_b1, nullptr, t16, (int)kBL, kD, kD, kD, kD, 1, kD, 1,0,0,0, 1);
  gemm(stream, t16, zow2, zo_b2, nullptr, eo16,   (int)kBL, kD, kD, kD, kD, 1, kD, 1,0,0,0, 0);
  k_kl_partial<<<dim3(256), dim3(256), 0, stream>>>(mean32, logv32, klpart, kBLD);
  k_kl_final<<<dim3(1), dim3(256), 0, stream>>>(klpart, out_kl, 256, 1.f / (float)kBLD);
  k_copy<<<dim3(gBLD), dim3(256), 0, stream>>>(mean32, out_z, kBLD);

  // === decoder ===
  k_build_xdec<<<dim3((unsigned)((nBLC + 255) / 256)), dim3(256), 0, stream>>>(xn32, xd32);
  k_embed<<<dim3(gBLD), dim3(256), 0, stream>>>(xd32, x_mark_dec, embd_tok, embd_temp, dv32, dv16);
  // self attention (causal, no delta)
  gemm(stream, dv16, swq, ds_bq, nullptr, q16,  (int)kBL, kD, kD, kD, kD, 1, kD, 1,0,0,0, 0);
  gemm(stream, dv16, swk, ds_bk, nullptr, kk16, (int)kBL, kD, kD, kD, kD, 1, kD, 1,0,0,0, 0);
  gemm(stream, dv16, swv, ds_bv, nullptr, vv16, (int)kBL, kD, kD, kD, kD, 1, kD, 1,0,0,0, 0);
  attention(stream, q16, kk16, vv16, scores, probs, ctx16, tau32, nullptr, /*causal=*/1);
  gemm(stream, ctx16, swo, ds_bo, tmp32, nullptr, (int)kBL, kD, kD, kD, kD, 1, kD, 1,0,0,0, 0);
  k_add_ln<<<dim3((unsigned)kBL), dim3(256), 0, stream>>>(dv32, tmp32, dl_g1, dl_b1n, dv32, dv16);
  // cross attention (kv = enc_out, with delta)
  gemm(stream, dv16, cwq, dc_bq, nullptr, q16,  (int)kBL, kD, kD, kD, kD, 1, kD, 1,0,0,0, 0);
  gemm(stream, eo16, cwk, dc_bk, nullptr, kk16, (int)kBL, kD, kD, kD, kD, 1, kD, 1,0,0,0, 0);
  gemm(stream, eo16, cwv, dc_bv, nullptr, vv16, (int)kBL, kD, kD, kD, kD, 1, kD, 1,0,0,0, 0);
  attention(stream, q16, kk16, vv16, scores, probs, ctx16, tau32, delta32, /*causal=*/0);
  gemm(stream, ctx16, cwo, dc_bo, tmp32, nullptr, (int)kBL, kD, kD, kD, kD, 1, kD, 1,0,0,0, 0);
  k_add_ln<<<dim3((unsigned)kBL), dim3(256), 0, stream>>>(dv32, tmp32, dl_g2, dl_b2n, dv32, dv16);
  // FFN
  gemm(stream, dv16,  dw1, dl_bf1, nullptr, mid16, (int)kBL, kF, kD, kD, kF, 1, kF, 1,0,0,0, 1);
  gemm(stream, mid16, dw2, dl_bf2, tmp32, nullptr, (int)kBL, kD, kF, kF, kD, 1, kD, 1,0,0,0, 0);
  k_add_ln<<<dim3((unsigned)kBL), dim3(256), 0, stream>>>(dv32, tmp32, dl_g3, dl_b3n, dv32, dv16);
  k_add_ln<<<dim3((unsigned)kBL), dim3(256), 0, stream>>>(dv32, nullptr, dec_g, dec_b, dv32, dv16);
  // final projection + de-normalization
  gemm(stream, dv16, pw16, proj_b, dp32, nullptr, (int)kBL, kC, kD, kD, kC, 1, kC, 1,0,0,0, 0);
  k_finalize<<<dim3((unsigned)((nBLC + 255) / 256)), dim3(256), 0, stream>>>(
      dp32, mean_s, std_s, out_pred, out_full);
}
